// GQA_61916248539454
// MI455X (gfx1250) — compile-verified, matrix operand order checked by reading, not run on hardware
//
// GQA block for MI455X / gfx1250.
// fp32 -> bf16 once; all GEMMs + attention through v_wmma_f32_16x16x32_bf16.
// All LDS tile staging (GEMM A/B tiles, attention K/Vt tiles) goes through the
// Tensor Data Mover (tensor_load_to_lds + s_wait_tensorcnt), double-buffered so
// DMA of tile i+1 overlaps WMMA on tile i. V is pre-transposed per-group in the
// V-projection epilogue so attention never transposes with VALU ops.
// ~155 GFLOP vs ~100 MB HBM traffic => compute bound; maximize WMMA density.

#include <hip/hip_runtime.h>
#include <hip/hip_bf16.h>

typedef __bf16 bf16;
typedef __attribute__((ext_vector_type(16))) __bf16 v16bf;
typedef __attribute__((ext_vector_type(8)))  float  v8f;
typedef __attribute__((ext_vector_type(4)))  unsigned int v4u;
typedef __attribute__((ext_vector_type(8)))  int v8i;
typedef __attribute__((ext_vector_type(4)))  int v4i;

#define D_MODEL 2048
#define NHEAD   32
#define GROUPS  8
#define HEAD_DIM 64
#define HPG     (NHEAD / GROUPS)       // 4
#define KV_DIM  (HEAD_DIM * GROUPS)    // 512
#define SEQ     2048
#define BATCH   2

#if defined(__AMDGCN__) && __has_builtin(__builtin_amdgcn_tensor_load_to_lds) && \
    __has_builtin(__builtin_amdgcn_s_wait_tensorcnt)
#define HAVE_TDM 1
#else
#define HAVE_TDM 0
#endif

// -------------------------------------------------------------------------
// fp32 -> bf16 conversion (memory bound, trivial at 23.3 TB/s)
// -------------------------------------------------------------------------
__global__ void f32_to_bf16_kernel(const float* __restrict__ in,
                                   bf16* __restrict__ out, int n) {
  for (int i = blockIdx.x * blockDim.x + threadIdx.x; i < n;
       i += gridDim.x * blockDim.x) {
    out[i] = (bf16)in[i];
  }
}

#if HAVE_TDM
// Issue one TDM 2D tile load: tile_rows x tile_cols bf16 tile from a row-major
// global tensor with the given row stride (elements), into contiguous LDS
// [tile_rows][tile_cols].
__device__ __forceinline__ void tdm_load_tile(const bf16* gptr, unsigned lds_off,
                                              unsigned tile_cols, unsigned tile_rows,
                                              unsigned long long row_stride) {
  unsigned long long ga = (unsigned long long)(uintptr_t)gptr;
  unsigned td0 = (unsigned)row_stride;   // tensor_dim0 (OOB bound), >= tile_cols
  unsigned td1 = tile_rows;
  v4u g0;
  g0.x = 1u;                                            // count=1, user D#
  g0.y = lds_off;                                       // lds_addr (bytes)
  g0.z = (unsigned)(ga & 0xffffffffu);                  // global_addr[31:0]
  g0.w = (unsigned)((ga >> 32) & 0x1ffffffu) | (2u << 30);  // [56:32] | type=2
  v8i g1;
  g1[0] = (int)(1u << 16);                              // data_size = 2 bytes
  g1[1] = (int)((td0 & 0xffffu) << 16);                 // tensor_dim0 lo16
  g1[2] = (int)(((td0 >> 16) & 0xffffu) | ((td1 & 0xffffu) << 16));  // d0 hi|d1 lo
  g1[3] = (int)(((td1 >> 16) & 0xffffu) | ((tile_cols & 0xffffu) << 16)); // |tile_d0
  g1[4] = (int)(tile_rows & 0xffffu);                   // tile_dim1 (tile_dim2=0)
  g1[5] = (int)(unsigned)(row_stride & 0xffffffffu);    // dim0 stride lo32 (elems)
  g1[6] = (int)(unsigned)((row_stride >> 32) & 0xffffu);
  g1[7] = 0;
  v4i z4 = {};
#if __clang_major__ >= 23
  v8i z8 = {};
  __builtin_amdgcn_tensor_load_to_lds(g0, g1, z4, z4, z8, 0);
#else
  __builtin_amdgcn_tensor_load_to_lds(g0, g1, z4, z4, 0);
#endif
}
#endif

// -------------------------------------------------------------------------
// bf16 WMMA GEMM:  C[M,N] = A[M,K] @ W[N,K]^T + bias
//   mode 0: f32 store              mode 1: bf16 store
//   mode 2: bf16 + fused RoPE      mode 3: bf16 transposed per 64-col group
// Block: 256 threads = 8 waves, block tile 128(M) x 128(N), K-step 32.
// Waves arranged 4(M) x 2(N); wave tile 32 x 64 -> 8 WMMA per K-step with
// B-fragment reuse across the two M-subtiles.
// TDM double-buffers the A/B LDS tiles: DMA(chunk i+1) overlaps WMMA(chunk i).
// -------------------------------------------------------------------------
__global__ __launch_bounds__(256)
void gemm_bf16_wmma(const bf16* __restrict__ A, const bf16* __restrict__ W,
                    const float* __restrict__ bias,
                    float* __restrict__ outF, bf16* __restrict__ outB,
                    int M, int N, int K, int mode, int S) {
  __shared__ bf16 ldsA[2][128 * 32];
  __shared__ bf16 ldsB[2][128 * 32];

  const int t    = threadIdx.x;
  const int lane = t & 31;
  const int wave = t >> 5;
  const int wm   = wave >> 1;         // 0..3  (M direction, 32 rows each)
  const int wn   = wave & 1;          // 0..1  (N direction, 64 cols each)
  const int m0   = blockIdx.x * 128;
  const int n0   = blockIdx.y * 128;

  v8f zero = {};
  v8f acc[2][4];
  for (int mt = 0; mt < 2; ++mt)
    for (int nt = 0; nt < 4; ++nt) acc[mt][nt] = zero;

  const int arow  = lane & 15;
  const int khalf = (lane >> 4) * 16;
  const int nk    = K / 32;

#if HAVE_TDM
  const unsigned ldsA0 = (unsigned)(uintptr_t)&ldsA[0][0];
  const unsigned ldsA1 = (unsigned)(uintptr_t)&ldsA[1][0];
  const unsigned ldsB0 = (unsigned)(uintptr_t)&ldsB[0][0];
  const unsigned ldsB1 = (unsigned)(uintptr_t)&ldsB[1][0];
  if (wave == 0) {
    tdm_load_tile(&A[(size_t)m0 * K], ldsA0, 32, 128, (unsigned)K);
    tdm_load_tile(&W[(size_t)n0 * K], ldsB0, 32, 128, (unsigned)K);
  }
  for (int ik = 0; ik < nk; ++ik) {
    const int cur = ik & 1;
    if (wave == 0) {
      if (ik + 1 < nk) {   // prefetch next chunk into the other buffer
        const int k1 = (ik + 1) * 32;
        tdm_load_tile(&A[(size_t)m0 * K + k1], cur ? ldsA0 : ldsA1, 32, 128, (unsigned)K);
        tdm_load_tile(&W[(size_t)n0 * K + k1], cur ? ldsB0 : ldsB1, 32, 128, (unsigned)K);
        __builtin_amdgcn_s_wait_tensorcnt(2);  // pair for chunk ik complete
      } else {
        __builtin_amdgcn_s_wait_tensorcnt(0);
      }
    }
    __syncthreads();
    v16bf a0 = *(const v16bf*)&ldsA[cur][(wm * 32 + arow) * 32 + khalf];
    v16bf a1 = *(const v16bf*)&ldsA[cur][(wm * 32 + 16 + arow) * 32 + khalf];
    for (int nt = 0; nt < 4; ++nt) {
      v16bf b = *(const v16bf*)&ldsB[cur][(wn * 64 + nt * 16 + arow) * 32 + khalf];
      acc[0][nt] = __builtin_amdgcn_wmma_f32_16x16x32_bf16(
          false, a0, false, b, (short)0, acc[0][nt], false, false);
      acc[1][nt] = __builtin_amdgcn_wmma_f32_16x16x32_bf16(
          false, a1, false, b, (short)0, acc[1][nt], false, false);
    }
    __syncthreads();
  }
#else
  // Fallback: cooperative VALU copy, single-buffered.
  for (int ik = 0; ik < nk; ++ik) {
    const int k0 = ik * 32;
    if (ik + 1 < nk) {
      __builtin_prefetch(&A[(size_t)(m0 + (t >> 1)) * K + k0 + 32], 0, 1);
      __builtin_prefetch(&W[(size_t)(n0 + (t >> 1)) * K + k0 + 32], 0, 1);
    }
    {
      int e = t * 16, row = e >> 5, kk = e & 31;
      const uint4* srcA = (const uint4*)&A[(size_t)(m0 + row) * K + k0 + kk];
      uint4* dstA = (uint4*)&ldsA[0][row * 32 + kk];
      dstA[0] = srcA[0]; dstA[1] = srcA[1];
      const uint4* srcB = (const uint4*)&W[(size_t)(n0 + row) * K + k0 + kk];
      uint4* dstB = (uint4*)&ldsB[0][row * 32 + kk];
      dstB[0] = srcB[0]; dstB[1] = srcB[1];
    }
    __syncthreads();
    v16bf a0 = *(const v16bf*)&ldsA[0][(wm * 32 + arow) * 32 + khalf];
    v16bf a1 = *(const v16bf*)&ldsA[0][(wm * 32 + 16 + arow) * 32 + khalf];
    for (int nt = 0; nt < 4; ++nt) {
      v16bf b = *(const v16bf*)&ldsB[0][(wn * 64 + nt * 16 + arow) * 32 + khalf];
      acc[0][nt] = __builtin_amdgcn_wmma_f32_16x16x32_bf16(
          false, a0, false, b, (short)0, acc[0][nt], false, false);
      acc[1][nt] = __builtin_amdgcn_wmma_f32_16x16x32_bf16(
          false, a1, false, b, (short)0, acc[1][nt], false, false);
    }
    __syncthreads();
  }
#endif

  // --------------------------- epilogue ---------------------------
  const int rbase = (lane < 16) ? 0 : 8;   // C layout: rows i / i+8
  const int col   = lane & 15;
  const int nwave = n0 + wn * 64;          // wave owns one full 64-dim head

  for (int mt = 0; mt < 2; ++mt) {
    const int mbase = m0 + wm * 32 + mt * 16;
    if (mode == 2) {
      // fused rotate-half RoPE: angle = pos * 10000^(-(d%32)/32)
      const float kLn1e4Over32 = 0.28782313662425572f;  // ln(10000)/32
      for (int tp = 0; tp < 2; ++tp) {
        int jlow = tp * 16 + col;            // d in [0,32)
        float theta = __expf(-(float)jlow * kLn1e4Over32);
        float blo = bias[nwave + jlow];
        float bhi = bias[nwave + jlow + 32];
        for (int i = 0; i < 8; ++i) {
          int m = mbase + rbase + i;
          int s = m % S;
          float a = acc[mt][tp][i]     + blo;
          float b = acc[mt][tp + 2][i] + bhi;
          float sn, cs;
          __sincosf((float)s * theta, &sn, &cs);
          outB[(size_t)m * N + nwave + jlow]      = (bf16)(a * cs - b * sn);
          outB[(size_t)m * N + nwave + jlow + 32] = (bf16)(b * cs + a * sn);
        }
      }
    } else if (mode == 3) {
      // V path: store transposed per 64-wide group: Vt[(group*64+d)*S + s]
      for (int nt = 0; nt < 4; ++nt)
        for (int i = 0; i < 8; ++i) {
          int m = mbase + rbase + i;
          int n = nwave + nt * 16 + col;
          int bb = m / S, s = m % S;
          int grp = n >> 6, d = n & 63;
          size_t gi = ((size_t)(bb * GROUPS + grp) * HEAD_DIM + d) * (size_t)S + s;
          outB[gi] = (bf16)(acc[mt][nt][i] + bias[n]);
        }
    } else if (mode == 1) {
      for (int nt = 0; nt < 4; ++nt)
        for (int i = 0; i < 8; ++i) {
          int m = mbase + rbase + i;
          int n = nwave + nt * 16 + col;
          outB[(size_t)m * N + n] = (bf16)(acc[mt][nt][i] + bias[n]);
        }
    } else {
      for (int nt = 0; nt < 4; ++nt)
        for (int i = 0; i < 8; ++i) {
          int m = mbase + rbase + i;
          int n = nwave + nt * 16 + col;
          outF[(size_t)m * N + n] = acc[mt][nt][i] + bias[n];
        }
    }
  }
}

// -------------------------------------------------------------------------
// Flash-style causal GQA attention.
// One block = 128 threads = 4 waves = one (b, head, 64-row query block).
// Each wave owns 16 query rows; loops over 64-key tiles with online softmax.
// K and pre-transposed Vt tiles are staged by the TDM (double-buffered), so
// the wave's issue slots are WMMA + softmax only.
// -------------------------------------------------------------------------
__global__ __launch_bounds__(128)
void gqa_flash_kernel(const bf16* __restrict__ Q, const bf16* __restrict__ Kk,
                      const bf16* __restrict__ Vt, bf16* __restrict__ O) {
  __shared__ bf16 ldsK[2][64 * 64];   // [key][d]
  __shared__ bf16 ldsVt[2][64 * 64];  // [d][key] (pre-transposed in global)
  __shared__ bf16 ldsP[4 * 16 * 64];  // per-wave P re-striping buffer

  const int t    = threadIdx.x;
  const int lane = t & 31;
  const int wave = t >> 5;

  int bid = blockIdx.x;
  const int nqb  = SEQ / 64;
  const int qblk = bid % nqb;  bid /= nqb;
  const int h    = bid % NHEAD; bid /= NHEAD;
  const int b    = bid;
  const int g    = h / HPG;
  const int q0   = qblk * 64;

  const int arow  = lane & 15;
  const int khalf = (lane >> 4) * 16;
  const int rbase = (lane < 16) ? 0 : 8;
  const int col   = lane & 15;

  const bf16* kbase  = &Kk[((size_t)(b * SEQ)) * KV_DIM + g * HEAD_DIM];
  const bf16* vtbase = &Vt[((size_t)(b * GROUPS + g)) * HEAD_DIM * SEQ];

  // Q fragments (16 rows x 64 dims = 2 A-frags), resident for whole loop
  const int qrow = q0 + wave * 16 + arow;
  const bf16* qptr = &Q[((size_t)(b * SEQ + qrow)) * D_MODEL + h * HEAD_DIM];
  v16bf aq0 = *(const v16bf*)&qptr[khalf];
  v16bf aq1 = *(const v16bf*)&qptr[32 + khalf];

  v8f zero = {};
  v8f accO[4];
  for (int i = 0; i < 4; ++i) accO[i] = zero;
  float rmax[8], rsum[8];
  for (int i = 0; i < 8; ++i) { rmax[i] = -3.0e38f; rsum[i] = 0.0f; }

  const float scale = 0.125f;  // 1/sqrt(64)

#if HAVE_TDM
  const unsigned lK0 = (unsigned)(uintptr_t)&ldsK[0][0];
  const unsigned lK1 = (unsigned)(uintptr_t)&ldsK[1][0];
  const unsigned lV0 = (unsigned)(uintptr_t)&ldsVt[0][0];
  const unsigned lV1 = (unsigned)(uintptr_t)&ldsVt[1][0];
  if (wave == 0) {
    tdm_load_tile(kbase, lK0, 64, 64, KV_DIM);
    tdm_load_tile(vtbase, lV0, 64, 64, SEQ);
  }
#endif

  for (int kb = 0; kb <= qblk; ++kb) {
    const int cur = kb & 1;
#if HAVE_TDM
    if (wave == 0) {
      if (kb + 1 <= qblk) {  // DMA next key block into the other buffer
        tdm_load_tile(kbase + (size_t)(kb + 1) * 64 * KV_DIM,
                      cur ? lK0 : lK1, 64, 64, KV_DIM);
        tdm_load_tile(vtbase + (kb + 1) * 64, cur ? lV0 : lV1, 64, 64, SEQ);
        __builtin_amdgcn_s_wait_tensorcnt(2);  // pair for block kb complete
      } else {
        __builtin_amdgcn_s_wait_tensorcnt(0);
      }
    }
    __syncthreads();
#else
    __syncthreads();
    {   // K tile [64 keys][64 d] and Vt tile [64 d][64 keys]: 4 x 16B each
      int row  = t >> 1;
      int half = (t & 1) * 32;
      const uint4* srcK = (const uint4*)
          &kbase[((size_t)(kb * 64 + row)) * KV_DIM + half];
      uint4* dstK = (uint4*)&ldsK[cur][row * 64 + half];
      dstK[0] = srcK[0]; dstK[1] = srcK[1]; dstK[2] = srcK[2]; dstK[3] = srcK[3];
      const uint4* srcV = (const uint4*)
          &vtbase[(size_t)row * SEQ + kb * 64 + half];
      uint4* dstV = (uint4*)&ldsVt[cur][row * 64 + half];
      dstV[0] = srcV[0]; dstV[1] = srcV[1]; dstV[2] = srcV[2]; dstV[3] = srcV[3];
    }
    __syncthreads();
#endif

    // ----- scores: 16 rows x 64 keys -----
    v8f sc[4];
    for (int nt = 0; nt < 4; ++nt) {
      v16bf bk0 = *(const v16bf*)&ldsK[cur][(nt * 16 + arow) * 64 + khalf];
      v16bf bk1 = *(const v16bf*)&ldsK[cur][(nt * 16 + arow) * 64 + 32 + khalf];
      v8f s = zero;
      s = __builtin_amdgcn_wmma_f32_16x16x32_bf16(false, aq0, false, bk0,
                                                  (short)0, s, false, false);
      s = __builtin_amdgcn_wmma_f32_16x16x32_bf16(false, aq1, false, bk1,
                                                  (short)0, s, false, false);
      sc[nt] = s;
    }

    // scale + causal mask (diagonal block only)
    for (int nt = 0; nt < 4; ++nt)
      for (int i = 0; i < 8; ++i) {
        float v = sc[nt][i] * scale;
        if (kb == qblk) {
          int key = kb * 64 + nt * 16 + col;
          int qr  = q0 + wave * 16 + rbase + i;
          if (key > qr) v = -3.0e38f;
        }
        sc[nt][i] = v;
      }

    // row-max across 64 keys (4 tiles + 16-lane shuffle group)
    float pm[8];
    for (int i = 0; i < 8; ++i) {
      float m = fmaxf(fmaxf(sc[0][i], sc[1][i]), fmaxf(sc[2][i], sc[3][i]));
      for (int off = 8; off >= 1; off >>= 1)
        m = fmaxf(m, __shfl_xor(m, off, 32));
      pm[i] = m;
    }
    float rescale[8];
    for (int i = 0; i < 8; ++i) {
      float mnew = fmaxf(rmax[i], pm[i]);
      rescale[i] = __expf(rmax[i] - mnew);
      rmax[i] = mnew;
    }

    // P = exp(s - m): accumulate row sums, re-stripe via per-wave LDS
    bf16* pw = &ldsP[wave * 16 * 64];
    float psum[8];
    for (int i = 0; i < 8; ++i) psum[i] = 0.0f;
    for (int nt = 0; nt < 4; ++nt)
      for (int i = 0; i < 8; ++i) {
        float p = __expf(sc[nt][i] - rmax[i]);
        psum[i] += p;
        pw[(rbase + i) * 64 + nt * 16 + col] = (bf16)p;
      }
    for (int i = 0; i < 8; ++i) {
      float ps = psum[i];
      for (int off = 8; off >= 1; off >>= 1) ps += __shfl_xor(ps, off, 32);
      rsum[i] = rsum[i] * rescale[i] + ps;
    }

    // rescale running O, then O += P @ V
    for (int dt = 0; dt < 4; ++dt)
      for (int i = 0; i < 8; ++i) accO[dt][i] = accO[dt][i] * rescale[i];

    v16bf ap0 = *(const v16bf*)&pw[arow * 64 + khalf];
    v16bf ap1 = *(const v16bf*)&pw[arow * 64 + 32 + khalf];
    for (int dt = 0; dt < 4; ++dt) {
      v16bf bv0 = *(const v16bf*)&ldsVt[cur][(dt * 16 + arow) * 64 + khalf];
      v16bf bv1 = *(const v16bf*)&ldsVt[cur][(dt * 16 + arow) * 64 + 32 + khalf];
      accO[dt] = __builtin_amdgcn_wmma_f32_16x16x32_bf16(
          false, ap0, false, bv0, (short)0, accO[dt], false, false);
      accO[dt] = __builtin_amdgcn_wmma_f32_16x16x32_bf16(
          false, ap1, false, bv1, (short)0, accO[dt], false, false);
    }
#if HAVE_TDM
    __syncthreads();   // tile consumed; TDM may overwrite this buffer next iter
#endif
  }

  // finalize: O / rowsum, store bf16 into [B*S, D_MODEL] layout
  for (int dt = 0; dt < 4; ++dt)
    for (int i = 0; i < 8; ++i) {
      int qr = q0 + wave * 16 + rbase + i;
      float o = accO[dt][i] / rsum[i];
      O[((size_t)(b * SEQ + qr)) * D_MODEL + h * HEAD_DIM + dt * 16 + col] =
          (bf16)o;
    }
}

// -------------------------------------------------------------------------
// Host-side orchestration (all on `stream`, graph-capture safe)
// -------------------------------------------------------------------------
extern "C" void kernel_launch(void* const* d_in, const int* in_sizes, int n_in,
                              void* d_out, int out_size, void* d_ws,
                              size_t ws_size, hipStream_t stream) {
  const float* x  = (const float*)d_in[0];
  const float* Wq = (const float*)d_in[1];
  const float* bq = (const float*)d_in[2];
  const float* Wk = (const float*)d_in[3];
  const float* bk = (const float*)d_in[4];
  const float* Wv = (const float*)d_in[5];
  const float* bv = (const float*)d_in[6];
  const float* Wo = (const float*)d_in[7];
  const float* bo = (const float*)d_in[8];
  float* out = (float*)d_out;

  const int M = BATCH * SEQ;  // 4096 tokens

  char* ws = (char*)d_ws;
  size_t off = 0;
  auto alloc = [&](size_t bytes) {
    char* p = ws + off;
    off += (bytes + 255) & ~(size_t)255;
    return p;
  };
  bf16* xb  = (bf16*)alloc((size_t)M * D_MODEL * 2);
  bf16* wqb = (bf16*)alloc((size_t)D_MODEL * D_MODEL * 2);
  bf16* wkb = (bf16*)alloc((size_t)KV_DIM * D_MODEL * 2);
  bf16* wvb = (bf16*)alloc((size_t)KV_DIM * D_MODEL * 2);
  bf16* wob = (bf16*)alloc((size_t)D_MODEL * D_MODEL * 2);
  bf16* qb  = (bf16*)alloc((size_t)M * D_MODEL * 2);
  bf16* kbuf= (bf16*)alloc((size_t)M * KV_DIM * 2);
  bf16* vtb = (bf16*)alloc((size_t)M * KV_DIM * 2);   // V pre-transposed
  bf16* ab  = (bf16*)alloc((size_t)M * D_MODEL * 2);

  auto cvt = [&](const float* src, bf16* dst, size_t n) {
    f32_to_bf16_kernel<<<1024, 256, 0, stream>>>(src, dst, (int)n);
  };
  cvt(x,  xb,  (size_t)M * D_MODEL);
  cvt(Wq, wqb, (size_t)D_MODEL * D_MODEL);
  cvt(Wk, wkb, (size_t)KV_DIM * D_MODEL);
  cvt(Wv, wvb, (size_t)KV_DIM * D_MODEL);
  cvt(Wo, wob, (size_t)D_MODEL * D_MODEL);

  // Q/K projections with fused RoPE epilogue; V with transposed store.
  gemm_bf16_wmma<<<dim3(M / 128, D_MODEL / 128), 256, 0, stream>>>(
      xb, wqb, bq, nullptr, qb, M, D_MODEL, D_MODEL, /*mode=*/2, SEQ);
  gemm_bf16_wmma<<<dim3(M / 128, KV_DIM / 128), 256, 0, stream>>>(
      xb, wkb, bk, nullptr, kbuf, M, KV_DIM, D_MODEL, /*mode=*/2, SEQ);
  gemm_bf16_wmma<<<dim3(M / 128, KV_DIM / 128), 256, 0, stream>>>(
      xb, wvb, bv, nullptr, vtb, M, KV_DIM, D_MODEL, /*mode=*/3, SEQ);

  // Flash causal GQA attention (TDM-staged K/Vt tiles).
  gqa_flash_kernel<<<BATCH * NHEAD * (SEQ / 64), 128, 0, stream>>>(
      qb, kbuf, vtb, ab);

  // Output projection, fp32 result.
  gemm_bf16_wmma<<<dim3(M / 128, D_MODEL / 128), 256, 0, stream>>>(
      ab, wob, bo, out, nullptr, M, D_MODEL, D_MODEL, /*mode=*/0, SEQ);
}
